// Attention_53111565582374
// MI455X (gfx1250) — matrix operand order repaired and checked by reading, not compiled, not run on hardware
//
#include <hip/hip_runtime.h>
#include <hip/hip_bf16.h>
#include <stdint.h>

typedef __attribute__((ext_vector_type(16))) _Float16 v16h;
typedef __attribute__((ext_vector_type(8)))  float    v8f;

#define S_LEN  2048
#define DHEAD  128
#define NH     16
#define NKV    4
#define BM     64          // query rows per workgroup (16 per wave)
#define BN     32          // key columns per iteration
#define NWAVES 4
// 1/sqrt(128) * log2(e): softmax done in exp2 domain (monotonic -> mask/max unchanged)
#define QK_SCALE2 (0.08838834764831845f * 1.4426950408889634f)

// ---------------------------------------------------------------------------
// LDS helpers: raw 32-bit LDS offset + async global->LDS 16B copy (ASYNCcnt)
// ---------------------------------------------------------------------------
typedef __attribute__((address_space(3))) char lds_char;

__device__ __forceinline__ unsigned lds_off(void* p) {
    return (unsigned)(uintptr_t)(lds_char*)p;       // addrspacecast generic->LDS, 32-bit offset
}

__device__ __forceinline__ void async_cp16(unsigned lds, const void* g) {
    unsigned long long ga = (unsigned long long)(uintptr_t)g;
    asm volatile("global_load_async_to_lds_b128 %0, %1, off"
                 :: "v"(lds), "v"(ga) : "memory");
}

__device__ __forceinline__ void wait_async0() {
    asm volatile("s_wait_asynccnt 0x0" ::: "memory");
}

// ---------------------------------------------------------------------------
// Pre-pass: fp32 K/V -> f16 staging buffers (K row-major, V transposed [d][s])
// ---------------------------------------------------------------------------
__global__ __launch_bounds__(256)
void cvt_kv_kernel(const float* __restrict__ kg, const float* __restrict__ vg,
                   _Float16* __restrict__ kh, _Float16* __restrict__ vt)
{
    const size_t idx = (size_t)blockIdx.x * 256 + threadIdx.x;  // over 2*4*2048*128
    const int d  = (int)(idx & (DHEAD - 1));
    const int s  = (int)((idx >> 7) & (S_LEN - 1));
    const size_t bh = idx >> 18;                                // 0..7
    kh[idx] = (_Float16)kg[idx];
    vt[bh * (size_t)(DHEAD * S_LEN) + (size_t)d * S_LEN + s] = (_Float16)vg[idx];
}

// ---------------------------------------------------------------------------
// Online-softmax update on the WMMA C layout (row = half*8+r, col = lane%16)
// ---------------------------------------------------------------------------
template<bool MASK>
__device__ __forceinline__ void softmax_update(
    v8f (&sacc)[2], v8f (&o)[8], float (&mrow)[8], float (&lrow)[8],
    _Float16 (*PbW)[BN], int kb, int qm0, int wave, int lh, int half)
{
    #pragma unroll
    for (int r = 0; r < 8; ++r) {
        if (MASK) {
            const int myrow = qm0 + wave * 16 + half * 8 + r;
            #pragma unroll
            for (int n = 0; n < 2; ++n) {
                const int kcol = kb * BN + n * 16 + lh;
                if (kcol > myrow) sacc[n][r] = -1e30f;
            }
        }
        float t = fmaxf(sacc[0][r], sacc[1][r]);
        #pragma unroll
        for (int xm = 1; xm < 16; xm <<= 1)
            t = fmaxf(t, __shfl_xor(t, xm, 16));     // reduce within lane-half
        const float mn    = fmaxf(mrow[r], t);
        const float alpha = exp2f(mrow[r] - mn);
        const float p0 = exp2f(sacc[0][r] - mn);
        const float p1 = exp2f(sacc[1][r] - mn);
        float ps = p0 + p1;
        #pragma unroll
        for (int xm = 1; xm < 16; xm <<= 1)
            ps += __shfl_xor(ps, xm, 16);
        lrow[r] = lrow[r] * alpha + ps;
        mrow[r] = mn;
        #pragma unroll
        for (int j = 0; j < 8; ++j) o[j][r] *= alpha;
        PbW[half * 8 + r][0  + lh] = (_Float16)p0;
        PbW[half * 8 + r][16 + lh] = (_Float16)p1;
    }
}

// ---------------------------------------------------------------------------
// Fast path: f16 staged K/V, double-buffered async-LDS pipeline
// ---------------------------------------------------------------------------
__global__ __launch_bounds__(128)
void fa_gqa_async_kernel(const float* __restrict__ qg,
                         const _Float16* __restrict__ kh,
                         const _Float16* __restrict__ vt,
                         float* __restrict__ outg)
{
    __shared__ __align__(16) _Float16 Kt[2][BN][DHEAD];      // 2 x 8 KB
    __shared__ __align__(16) _Float16 Vt[2][DHEAD][BN];      // 2 x 8 KB (transposed tile)
    __shared__ __align__(16) _Float16 Pb[NWAVES][16][BN];    // 4 KB

    const int tid  = threadIdx.x;
    const int lane = tid & 31;
    const int wave = tid >> 5;
    const int lh   = lane & 15;
    const int half = lane >> 4;

    const int mblk = blockIdx.x;
    const int h    = blockIdx.y;
    const int b    = blockIdx.z;
    const int hkv  = h >> 2;

    const int qm0 = mblk * BM;

    const float*    qbase  = qg + (((size_t)b * NH  + h  ) * S_LEN) * DHEAD;
    const _Float16* khbase = kh + ((size_t)(b * NKV + hkv)) * S_LEN * DHEAD;
    const _Float16* vtbase = vt + ((size_t)(b * NKV + hkv)) * DHEAD * S_LEN;

    const unsigned ktOff[2] = { lds_off(&Kt[0][0][0]), lds_off(&Kt[1][0][0]) };
    const unsigned vtOff[2] = { lds_off(&Vt[0][0][0]), lds_off(&Vt[1][0][0]) };

    // ---- Q rows -> A-fragments (M = lane%16; K = half*8 + (v&3)*2 + (v>>2)*16)
    const int qrow_frag = qm0 + wave * 16 + lh;
    v16h qa[4];
    #pragma unroll
    for (int f = 0; f < 4; ++f) {
        #pragma unroll
        for (int vv = 0; vv < 8; ++vv) {
            const int kk = f * 32 + half * 8 + (vv & 3) * 2 + (vv >> 2) * 16;
            const float2 pr = *(const float2*)(qbase + (size_t)qrow_frag * DHEAD + kk);
            qa[f][2 * vv]     = (_Float16)(pr.x * QK_SCALE2);
            qa[f][2 * vv + 1] = (_Float16)(pr.y * QK_SCALE2);
        }
    }

    v8f o[8];
    float mrow[8], lrow[8];
    #pragma unroll
    for (int j = 0; j < 8; ++j) o[j] = (v8f){0.f,0.f,0.f,0.f,0.f,0.f,0.f,0.f};
    #pragma unroll
    for (int r = 0; r < 8; ++r) { mrow[r] = -1e30f; lrow[r] = 0.f; }

    const int nkb = qm0 / BN + 2;

    // issue a K tile (8 KB contiguous) + V^T tile (128 rows x 64 B) into buffer
    auto issue_tile = [&](int kb, int buf) {
        const char* kgp = (const char*)(khbase + (size_t)kb * BN * DHEAD);
        const char* vgp = (const char*)vtbase + (size_t)kb * BN * 2;
        #pragma unroll
        for (int i = 0; i < 4; ++i) {
            const int c = tid + 128 * i;              // 0..511 16-byte chunks
            async_cp16(ktOff[buf] + c * 16, kgp + (size_t)c * 16);
            const int d = c >> 2, p = c & 3;
            async_cp16(vtOff[buf] + d * 64 + p * 16,
                       vgp + (size_t)d * (S_LEN * 2) + p * 16);
        }
    };

    issue_tile(0, 0);

    for (int kb = 0; kb < nkb; ++kb) {
        const int buf = kb & 1;
        wait_async0();        // own async writes for tile kb complete
        __syncthreads();      // all waves' writes complete; all done reading buf^1
        if (kb + 1 < nkb) issue_tile(kb + 1, buf ^ 1);   // prefetch under compute

        // ---- S = Q K^T
        v8f sacc[2];
        #pragma unroll
        for (int n = 0; n < 2; ++n) {
            sacc[n] = (v8f){0.f,0.f,0.f,0.f,0.f,0.f,0.f,0.f};
            const int scol = n * 16 + lh;
            #pragma unroll
            for (int d0 = 0; d0 < 4; ++d0) {
                v16h bf;
                #pragma unroll
                for (int vv = 0; vv < 8; ++vv) {
                    const int kd = d0 * 32 + half * 8 + (vv & 3) * 2 + (vv >> 2) * 16;
                    bf[2 * vv]     = Kt[buf][scol][kd];
                    bf[2 * vv + 1] = Kt[buf][scol][kd + 1];
                }
                sacc[n] = __builtin_amdgcn_wmma_f32_16x16x32_f16(
                    false, qa[d0], false, bf, (short)0, sacc[n], false, false);
            }
        }

        // ---- softmax (mask only on the <=2 diagonal blocks)
        if (kb < nkb - 2)
            softmax_update<false>(sacc, o, mrow, lrow, Pb[wave], kb, qm0, wave, lh, half);
        else
            softmax_update<true >(sacc, o, mrow, lrow, Pb[wave], kb, qm0, wave, lh, half);

        // ---- O += P @ V
        v16h pa;
        #pragma unroll
        for (int vv = 0; vv < 8; ++vv) {
            const int kd = half * 8 + (vv & 3) * 2 + (vv >> 2) * 16;
            pa[2 * vv]     = Pb[wave][lh][kd];
            pa[2 * vv + 1] = Pb[wave][lh][kd + 1];
        }
        #pragma unroll
        for (int j = 0; j < 8; ++j) {
            v16h vb;
            const int c = j * 16 + lh;
            #pragma unroll
            for (int vv = 0; vv < 8; ++vv) {
                const int ks = half * 8 + (vv & 3) * 2 + (vv >> 2) * 16;
                vb[2 * vv]     = Vt[buf][c][ks];
                vb[2 * vv + 1] = Vt[buf][c][ks + 1];
            }
            o[j] = __builtin_amdgcn_wmma_f32_16x16x32_f16(
                false, pa, false, vb, (short)0, o[j], false, false);
        }
    }

    #pragma unroll
    for (int r = 0; r < 8; ++r) {
        const int row   = qm0 + wave * 16 + half * 8 + r;
        const float inv = 1.0f / lrow[r];
        #pragma unroll
        for (int j = 0; j < 8; ++j) {
            const int col = h * DHEAD + j * 16 + lh;
            outg[((size_t)b * S_LEN + row) * (NH * DHEAD) + col] = o[j][r] * inv;
        }
    }
}

// ---------------------------------------------------------------------------
// Fallback (self-contained, fp32 loads + in-kernel convert) if ws too small
// ---------------------------------------------------------------------------
__global__ __launch_bounds__(128)
void fa_gqa_fallback(const float* __restrict__ qg, const float* __restrict__ kg,
                     const float* __restrict__ vg, float* __restrict__ outg)
{
    __shared__ __align__(16) _Float16 Kt[BN][DHEAD];
    __shared__ __align__(16) _Float16 Vt[DHEAD][BN];
    __shared__ __align__(16) _Float16 Pb[NWAVES][16][BN];

    const int tid  = threadIdx.x;
    const int lane = tid & 31;
    const int wave = tid >> 5;
    const int lh   = lane & 15;
    const int half = lane >> 4;

    const int mblk = blockIdx.x, h = blockIdx.y, b = blockIdx.z;
    const int hkv  = h >> 2;
    const int qm0  = mblk * BM;

    const float* qbase = qg + (((size_t)b * NH  + h  ) * S_LEN) * DHEAD;
    const float* kbase = kg + (((size_t)b * NKV + hkv) * S_LEN) * DHEAD;
    const float* vbase = vg + (((size_t)b * NKV + hkv) * S_LEN) * DHEAD;

    const int qrow_frag = qm0 + wave * 16 + lh;
    v16h qa[4];
    #pragma unroll
    for (int f = 0; f < 4; ++f)
        #pragma unroll
        for (int vv = 0; vv < 8; ++vv) {
            const int kk = f * 32 + half * 8 + (vv & 3) * 2 + (vv >> 2) * 16;
            const float2 pr = *(const float2*)(qbase + (size_t)qrow_frag * DHEAD + kk);
            qa[f][2 * vv]     = (_Float16)(pr.x * QK_SCALE2);
            qa[f][2 * vv + 1] = (_Float16)(pr.y * QK_SCALE2);
        }

    v8f o[8];
    float mrow[8], lrow[8];
    #pragma unroll
    for (int j = 0; j < 8; ++j) o[j] = (v8f){0.f,0.f,0.f,0.f,0.f,0.f,0.f,0.f};
    #pragma unroll
    for (int r = 0; r < 8; ++r) { mrow[r] = -1e30f; lrow[r] = 0.f; }

    const int nkb = qm0 / BN + 2;
    for (int kb = 0; kb < nkb; ++kb) {
        {
            const int row = tid >> 2, cb = (tid & 3) * 32;
            const float* kp = kbase + (size_t)(kb * BN + row) * DHEAD + cb;
            const float* vp = vbase + (size_t)(kb * BN + row) * DHEAD + cb;
            #pragma unroll
            for (int i = 0; i < 32; i += 4) {
                const float4 k4 = *(const float4*)(kp + i);
                Kt[row][cb+i+0] = (_Float16)k4.x; Kt[row][cb+i+1] = (_Float16)k4.y;
                Kt[row][cb+i+2] = (_Float16)k4.z; Kt[row][cb+i+3] = (_Float16)k4.w;
                const float4 v4 = *(const float4*)(vp + i);
                Vt[cb+i+0][row] = (_Float16)v4.x; Vt[cb+i+1][row] = (_Float16)v4.y;
                Vt[cb+i+2][row] = (_Float16)v4.z; Vt[cb+i+3][row] = (_Float16)v4.w;
            }
        }
        __syncthreads();

        v8f sacc[2];
        #pragma unroll
        for (int n = 0; n < 2; ++n) {
            sacc[n] = (v8f){0.f,0.f,0.f,0.f,0.f,0.f,0.f,0.f};
            const int scol = n * 16 + lh;
            #pragma unroll
            for (int d0 = 0; d0 < 4; ++d0) {
                v16h bf;
                #pragma unroll
                for (int vv = 0; vv < 8; ++vv) {
                    const int kd = d0 * 32 + half * 8 + (vv & 3) * 2 + (vv >> 2) * 16;
                    bf[2*vv] = Kt[scol][kd]; bf[2*vv+1] = Kt[scol][kd+1];
                }
                sacc[n] = __builtin_amdgcn_wmma_f32_16x16x32_f16(
                    false, qa[d0], false, bf, (short)0, sacc[n], false, false);
            }
        }

        if (kb < nkb - 2)
            softmax_update<false>(sacc, o, mrow, lrow, Pb[wave], kb, qm0, wave, lh, half);
        else
            softmax_update<true >(sacc, o, mrow, lrow, Pb[wave], kb, qm0, wave, lh, half);

        v16h pa;
        #pragma unroll
        for (int vv = 0; vv < 8; ++vv) {
            const int kd = half * 8 + (vv & 3) * 2 + (vv >> 2) * 16;
            pa[2*vv] = Pb[wave][lh][kd]; pa[2*vv+1] = Pb[wave][lh][kd+1];
        }
        #pragma unroll
        for (int j = 0; j < 8; ++j) {
            v16h vb;
            const int c = j * 16 + lh;
            #pragma unroll
            for (int vv = 0; vv < 8; ++vv) {
                const int ks = half * 8 + (vv & 3) * 2 + (vv >> 2) * 16;
                vb[2*vv] = Vt[c][ks]; vb[2*vv+1] = Vt[c][ks+1];
            }
            o[j] = __builtin_amdgcn_wmma_f32_16x16x32_f16(
                false, pa, false, vb, (short)0, o[j], false, false);
        }
        __syncthreads();
    }

    #pragma unroll
    for (int r = 0; r < 8; ++r) {
        const int row = qm0 + wave * 16 + half * 8 + r;
        const float inv = 1.0f / lrow[r];
        #pragma unroll
        for (int j = 0; j < 8; ++j)
            outg[((size_t)b * S_LEN + row) * (NH * DHEAD) + h * DHEAD + j * 16 + lh]
                = o[j][r] * inv;
    }
}

extern "C" void kernel_launch(void* const* d_in, const int* in_sizes, int n_in,
                              void* d_out, int out_size, void* d_ws, size_t ws_size,
                              hipStream_t stream) {
    (void)in_sizes; (void)n_in; (void)out_size;
    const float* q = (const float*)d_in[0];
    const float* k = (const float*)d_in[1];
    const float* v = (const float*)d_in[2];
    float* out = (float*)d_out;

    const size_t kv_elems = (size_t)2 * NKV * S_LEN * DHEAD;      // 2,097,152
    const size_t need     = 2 * kv_elems * sizeof(_Float16);      // 8 MB

    dim3 grid(S_LEN / BM, NH, 2);
    dim3 block(128);

    if (ws_size >= need) {
        _Float16* kh = (_Float16*)d_ws;
        _Float16* vt = kh + kv_elems;
        cvt_kv_kernel<<<dim3((unsigned)(kv_elems / 256)), 256, 0, stream>>>(k, v, kh, vt);
        fa_gqa_async_kernel<<<grid, block, 0, stream>>>(q, kh, vt, out);
    } else {
        fa_gqa_fallback<<<grid, block, 0, stream>>>(q, k, v, out);
    }
}
